// RecurrentState_30992484008021
// MI455X (gfx1250) — matrix operand order repaired and checked by reading, not compiled, not run on hardware
//
#include <hip/hip_runtime.h>

// ---------------------------------------------------------------------------
// RecurrentState LSTM for MI455X (gfx1250, wave32, WMMA bf16)
//   B=16, T=2048, H=768 (12 heads x 64), 4H=3072
// Phase 0: convert W_ih / W_hh fp32 -> bf16 (row-major [3072][768])
// Phase 1: xg[T][B][3072] = x @ W_ih^T + (b_ih + b_hh)   (parallel GEMM,
//          software-pipelined double-buffered WMMA)
// Phase 2: persistent 16-WG scan: W_hh slice (288KB bf16) staged into LDS in
//          WMMA-fragment order (conflict-free ds_load_b128), double-buffered
//          A(global)/B(LDS) pipeline, c in VGPRs, global barrier per step.
// ---------------------------------------------------------------------------

typedef __attribute__((ext_vector_type(16))) __bf16 v16bf;
typedef __attribute__((ext_vector_type(8)))  float  v8f;

#define B_SZ   16
#define T_SZ   2048
#define H_SZ   768
#define G_SZ   3072          // 4*H
#define NHEAD  12
#define HD     64

#define NWG        16        // persistent workgroups in scan kernel
#define WG_DIMS    48        // h-dims owned per workgroup
#define WG_WAVES   3         // 16 dims per wave
#define WG_THREADS 96
#define NTILES     12        // WG_WAVES * 4 gates, 16x16 tiles
#define LDS_BYTES  (NTILES * 24 * 1024)   // 294912 B = 288KB (fits 320KB WGP)

typedef union { v16bf v; uint4 q[2]; unsigned short u[16]; } Frag;

__device__ __forceinline__ unsigned short f2bf(float x) {
  union { float f; unsigned u; } v; v.f = x;
  unsigned u = v.u;
  unsigned r = u + 0x7FFFu + ((u >> 16) & 1u);   // round-to-nearest-even
  return (unsigned short)(r >> 16);
}

// Fast activations using v_exp_f32 / v_rcp_f32 (TRANS ops co-execute with
// XDL WMMA per CDNA5 ISA). tanh form saturates cleanly at +/-1.
__device__ __forceinline__ float fast_rcp(float x) {
  return __builtin_amdgcn_rcpf(x);
}
__device__ __forceinline__ float sigmoidf_(float x) {
  return fast_rcp(1.0f + __expf(-x));
}
__device__ __forceinline__ float tanhf_(float x) {
  float e = __expf(2.0f * x);
  return 1.0f - 2.0f * fast_rcp(e + 1.0f);
}

__device__ __forceinline__ v8f wmma_bf16(const Frag& a, const Frag& b, v8f c) {
  return __builtin_amdgcn_wmma_f32_16x16x32_bf16(
      false, a.v, false, b.v, (short)0, c, false, false);
}

// ----------------------------- Phase 0 -------------------------------------
__global__ void conv_bf16_kernel(const float* __restrict__ src,
                                 unsigned short* __restrict__ dst, int n) {
  int i = blockIdx.x * blockDim.x + threadIdx.x;
  int stride = gridDim.x * blockDim.x;
  for (; i < n; i += stride) dst[i] = f2bf(src[i]);
}

// ----------------------------- Phase 1 -------------------------------------
// grid = (T, 12), block = 128 (4 waves). Wave computes a 16x64 tile:
// rows = 16 batch elements at time t, cols = 64 gate columns.
struct RawA { float4 x0, x1, x2, x3; };   // 16 fp32 of one A fragment

__global__ __launch_bounds__(128)
void xg_gemm_kernel(const float* __restrict__ state,          // [B,12,T,64]
                    const unsigned short* __restrict__ Wihb,  // [3072][768] bf16
                    const float* __restrict__ b_ih,
                    const float* __restrict__ b_hh,
                    float* __restrict__ xg) {                 // [T][B][3072]
  const int t      = blockIdx.x;
  const int colBlk = blockIdx.y;              // 256 cols per block
  const int wave   = threadIdx.x >> 5;
  const int lane   = threadIdx.x & 31;
  const int n      = lane & 15;               // tile column / A row (batch)
  const int half   = lane >> 4;               // lane group
  const int colW   = colBlk * 256 + wave * 64;

  // A-frag source: x[batch=n, t, k]; lane holds row n,
  // K in {k0+half*16..+15}
  auto loadRawA = [&](RawA& r, int kc) {
    const int k0  = kc * 32;
    const int kA0 = k0 + half * 16;
    const int kA1 = k0 + half * 16 + 8;
    const float* p0 = state + ((size_t)(n * NHEAD + (kA0 >> 6)) * T_SZ + t) * HD + (kA0 & 63);
    const float* p1 = state + ((size_t)(n * NHEAD + (kA1 >> 6)) * T_SZ + t) * HD + (kA1 & 63);
    r.x0 = ((const float4*)p0)[0];
    r.x1 = ((const float4*)p0)[1];
    r.x2 = ((const float4*)p1)[0];
    r.x3 = ((const float4*)p1)[1];
  };
  auto convA = [&](const RawA& r, Frag& f) {
    const float* s = (const float*)&r;
#pragma unroll
    for (int c = 0; c < 16; ++c) f.u[c] = f2bf(s[c]);
  };
  // B-frag: column nn=n of B equals W row (colW+g*16+n); lane holds 16
  // contiguous K values starting at k0+half*16 (32B contiguous load).
  auto loadB4 = [&](Frag* B, int kc) {
    const int k0 = kc * 32;
#pragma unroll
    for (int g = 0; g < 4; ++g) {
      const uint4* wb = (const uint4*)(Wihb + (size_t)(colW + g * 16 + n) * H_SZ
                                             + k0 + half * 16);
      B[g].q[0] = wb[0];
      B[g].q[1] = wb[1];
    }
  };

  v8f acc[4];
#pragma unroll
  for (int g = 0; g < 4; ++g)
#pragma unroll
    for (int r = 0; r < 8; ++r) acc[g][r] = 0.0f;

  RawA ra0, ra1;
  Frag A0, A1, B0[4], B1[4];
  loadRawA(ra0, 0);
  loadB4(B0, 0);

  for (int kc = 0; kc < 24; kc += 2) {
    const int kn1 = kc + 1;
    const int kn2 = (kc + 2 < 24) ? (kc + 2) : 0;   // clamped (unused at tail)
    loadRawA(ra1, kn1);
    loadB4(B1, kn1);
    convA(ra0, A0);
#pragma unroll
    for (int g = 0; g < 4; ++g) acc[g] = wmma_bf16(A0, B0[g], acc[g]);
    loadRawA(ra0, kn2);
    loadB4(B0, kn2);
    convA(ra1, A1);
#pragma unroll
    for (int g = 0; g < 4; ++g) acc[g] = wmma_bf16(A1, B1[g], acc[g]);
  }

  // store with bias; lane holds column (colW+g*16+n), rows half*8+r
#pragma unroll
  for (int g = 0; g < 4; ++g) {
    const int col = colW + g * 16 + n;
    const float bias = b_ih[col] + b_hh[col];
#pragma unroll
    for (int r = 0; r < 8; ++r) {
      const int m = half * 8 + r;
      xg[((size_t)(t * B_SZ + m)) * G_SZ + col] = acc[g][r] + bias;
    }
  }
}

// ----------------------------- Phase 2 -------------------------------------
// Persistent scan: grid = 16 WGs x 96 threads (3 waves). WG owns 48 h-dims;
// wave owns 16 dims -> 4 accumulators = i/f/g/o 16x16 tiles. The WG's W_hh
// slice (192 rows x 768, bf16 = 288KB) is staged once into LDS in
// WMMA-fragment order: block (tile tt, kc) = 1KB at offset ((tt*24+kc)<<10),
// lane's 32B at +lane*32  -> conflict-free sequential ds_load_b128.
__global__ __launch_bounds__(WG_THREADS)
void lstm_scan_kernel(const float* __restrict__ xg,            // [T][B][3072]
                      const unsigned short* __restrict__ Whhb, // [3072][768] bf16
                      unsigned short* __restrict__ hbuf,       // [2][16][768] bf16
                      float* __restrict__ y,                   // [B][12][T][64]
                      unsigned int* __restrict__ barCnt) {
  extern __shared__ unsigned char smem[];

  const int blk  = blockIdx.x;               // 0..15
  const int tid  = threadIdx.x;
  const int wave = tid >> 5;                 // 0..2
  const int lane = tid & 31;
  const int n    = lane & 15;                // batch row of A / col of tile
  const int half = lane >> 4;
  const int j0   = blk * WG_DIMS + wave * 16;  // this wave's h-dim base
  const int head = j0 >> 6;                    // 16-dim range never crosses head
  const int dcol = j0 & 63;

  // ---- stage W_hh slice into LDS in fragment order (once) ----
  // chunk c (32B): lane' = c&31, kc = (c>>5)%24, tile tt = c/(32*24);
  // tile tt: wave w = tt>>2, gate g = tt&3.
  for (int c = tid; c < NTILES * 24 * 32; c += WG_THREADS) {
    const int ln = c & 31;
    const int kc = (c >> 5) % 24;
    const int tt = c / (32 * 24);
    const int g  = tt & 3;
    const int w  = tt >> 2;
    const int row = g * H_SZ + blk * WG_DIMS + w * 16 + (ln & 15);
    const int kk  = kc * 32 + (ln >> 4) * 16;
    const uint4* src = (const uint4*)(Whhb + (size_t)row * H_SZ + kk);
    uint4* dst = (uint4*)smem + (size_t)c * 2;
    dst[0] = src[0];
    dst[1] = src[1];
  }
  __syncthreads();

  const uint4* WlQ = (const uint4*)smem;
  auto loadB4 = [&](Frag* B, int kc) {
#pragma unroll
    for (int g = 0; g < 4; ++g) {
      const uint4* p = WlQ + ((((wave * 4 + g) * 24 + kc) << 6) + lane * 2);
      B[g].q[0] = p[0];
      B[g].q[1] = p[1];
    }
  };

  float c[8];
#pragma unroll
  for (int r = 0; r < 8; ++r) c[r] = 0.0f;

  for (int t = 0; t < T_SZ; ++t) {
    const unsigned short* hc = hbuf + (size_t)(t & 1) * B_SZ * H_SZ;
    unsigned short*       hn = hbuf + (size_t)((t + 1) & 1) * B_SZ * H_SZ;

    auto loadA = [&](Frag& f, int kc) {
      const int k0 = kc * 32;
      f.q[0] = *(const uint4*)(hc + n * H_SZ + k0 + half * 16);
      f.q[1] = *(const uint4*)(hc + n * H_SZ + k0 + half * 16 + 8);
    };

    // init accumulators with xg slice (acts as the C operand); xg is read
    // exactly once -> non-temporal so L2 stays dedicated to h traffic
    v8f acc[4];
#pragma unroll
    for (int g = 0; g < 4; ++g)
#pragma unroll
      for (int r = 0; r < 8; ++r)
        acc[g][r] = __builtin_nontemporal_load(
            xg + ((size_t)(t * B_SZ + half * 8 + r)) * G_SZ + g * H_SZ + j0 + n);

    // double-buffered A(global h) / B(LDS W) pipeline over K = 24 chunks
    Frag A0, A1, B0[4], B1[4];
    loadA(A0, 0);
    loadB4(B0, 0);
    for (int kc = 0; kc < 24; kc += 2) {
      const int kn1 = kc + 1;
      const int kn2 = (kc + 2 < 24) ? (kc + 2) : 0;   // clamped (unused at tail)
      loadA(A1, kn1);
      loadB4(B1, kn1);
#pragma unroll
      for (int g = 0; g < 4; ++g) acc[g] = wmma_bf16(A0, B0[g], acc[g]);
      loadA(A0, kn2);
      loadB4(B0, kn2);
#pragma unroll
      for (int g = 0; g < 4; ++g) acc[g] = wmma_bf16(A1, B1[g], acc[g]);
    }

    // elementwise gate math; lane holds (batch m = half*8+r, dim j0+n)
#pragma unroll
    for (int r = 0; r < 8; ++r) {
      const int m = half * 8 + r;
      const float iv = sigmoidf_(acc[0][r]);
      const float fv = sigmoidf_(acc[1][r]);
      const float gv = tanhf_(acc[2][r]);
      const float ov = sigmoidf_(acc[3][r]);
      c[r] = fv * c[r] + iv * gv;
      const float hv = ov * tanhf_(c[r]);
      hn[m * H_SZ + j0 + n] = f2bf(hv);
      __builtin_nontemporal_store(
          hv, y + (((size_t)m * NHEAD + head) * T_SZ + t) * HD + dcol + n);
    }

    // prefetch next step's xg tile (independent of the barrier): 32 lanes x 2
    // prefetches cover all 64 cachelines of the wave's next [16 x 64] slice
    if (t + 1 < T_SZ) {
      const float* pre = xg + ((size_t)((t + 1) * B_SZ + n)) * G_SZ
                            + (half * 2) * H_SZ + j0;
      __builtin_prefetch(pre, 0, 0);
      __builtin_prefetch(pre + H_SZ, 0, 0);
    }

    // ---- device-wide step barrier (monotonic counter, reset per launch) ----
    __threadfence();
    __syncthreads();
    if (tid == 0) {
      atomicAdd(barCnt, 1u);
      const unsigned int target = (unsigned int)NWG * (unsigned int)(t + 1);
      while (atomicAdd(barCnt, 0u) < target) {
        __builtin_amdgcn_s_sleep(2);
      }
    }
    __syncthreads();
  }
}

// ---------------------------------------------------------------------------
extern "C" void kernel_launch(void* const* d_in, const int* in_sizes, int n_in,
                              void* d_out, int out_size, void* d_ws, size_t ws_size,
                              hipStream_t stream) {
  const float* state = (const float*)d_in[0];   // [16,12,2048,64]
  const float* W_ih  = (const float*)d_in[1];   // [3072,768]
  const float* W_hh  = (const float*)d_in[2];   // [3072,768]
  const float* b_ih  = (const float*)d_in[3];   // [3072]
  const float* b_hh  = (const float*)d_in[4];   // [3072]
  float* y = (float*)d_out;                     // [16,12,2048,64]

  char* ws = (char*)d_ws;
  const size_t nW     = (size_t)G_SZ * H_SZ;            // 3072*768
  const size_t offWih = 0;
  const size_t offWhh = offWih + nW * sizeof(unsigned short);
  const size_t offH   = offWhh + nW * sizeof(unsigned short);
  const size_t szH    = (size_t)2 * B_SZ * H_SZ * sizeof(unsigned short);
  const size_t offBar = offH + szH;
  const size_t offXg  = offBar + 256;

  unsigned short* Wihb = (unsigned short*)(ws + offWih);
  unsigned short* Whhb = (unsigned short*)(ws + offWhh);
  unsigned short* hbuf = (unsigned short*)(ws + offH);
  unsigned int*   bar  = (unsigned int*)(ws + offBar);
  float*          xg   = (float*)(ws + offXg);

  // zero h ping-pong buffers + barrier counter (graph-capture safe memset)
  hipMemsetAsync(ws + offH, 0, szH + 256, stream);

  // Phase 0: weight conversion fp32 -> bf16
  conv_bf16_kernel<<<1024, 256, 0, stream>>>(W_ih, Wihb, (int)nW);
  conv_bf16_kernel<<<1024, 256, 0, stream>>>(W_hh, Whhb, (int)nW);

  // Phase 1: xg = x @ W_ih^T + (b_ih + b_hh)
  xg_gemm_kernel<<<dim3(T_SZ, G_SZ / 256), 128, 0, stream>>>(
      state, Wihb, b_ih, b_hh, xg);

  // Phase 2: persistent recurrent scan (288KB dynamic LDS per WG)
  lstm_scan_kernel<<<NWG, WG_THREADS, LDS_BYTES, stream>>>(
      xg, Whhb, hbuf, y, bar);
}
